// GraphNetworkGNLayer_25598005084727
// MI455X (gfx1250) — matrix operand order repaired and checked by reading, not compiled, not run on hardware
//
#include <hip/hip_runtime.h>

// GN layer for MI455X (gfx1250): bf16 WMMA (f32 accum) MLPs.
// v2: MT=2 M-tiles per wave (halves L2 weight traffic, 2 WMMAs per B load),
// A-fragments built directly from global f32 (no input LDS staging).

typedef __attribute__((ext_vector_type(16))) __bf16 v16bf;
typedef __attribute__((ext_vector_type(8)))  __bf16 v8bf;
typedef __attribute__((ext_vector_type(8)))  float  v8f;
typedef __attribute__((ext_vector_type(4)))  float  v4f;

#define N_NODES 50000
#define N_EDGES 800000
#define FDIM 64
#define LAT 128
#define MT 2  // M-tiles (of 16 rows) per wave

#define NODE_TILES (N_NODES / 16)                 // 3125
#define EDGE_SUPER (N_EDGES / (16 * MT))          // 25000 (exact)
#define EDGE_BLOCKS (EDGE_SUPER / 4)              // 6250 (exact)
#define NODE_SUPER ((NODE_TILES + MT - 1) / MT)   // 1563
#define NODE_BLOCKS ((NODE_SUPER + 3) / 4)        // 391

static __device__ __forceinline__ v8f wmma_bf16(v16bf a, v16bf b, v8f c) {
  return __builtin_amdgcn_wmma_f32_16x16x32_bf16(false, a, false, b,
                                                 (short)0, c, false, false);
}

// Intra-wave LDS write->read ordering fence (DS ops are in-order per wave).
#define LDS_FENCE() asm volatile("s_wait_dscnt 0x0" ::: "memory")

// Build a 16x32 bf16 A-fragment directly from a per-lane f32 row pointer.
// Lane l holds row m = l&15; element i<8 -> feature off+i, i>=8 -> off+16+(i-8)
// with off = (kt&1)*32 + (l>>4)*8 (matches ISA 16-bit A layout).
template <bool SCALE>
static __device__ __forceinline__ v16bf a_frag_from_f32(
    const float* __restrict__ p, int off, float s) {
  v4f f0 = *(const v4f*)(p + off);
  v4f f1 = *(const v4f*)(p + off + 4);
  v4f f2 = *(const v4f*)(p + off + 16);
  v4f f3 = *(const v4f*)(p + off + 20);
  if (SCALE) { f0 *= s; f1 *= s; f2 *= s; f3 *= s; }
  v16bf a;
#pragma unroll
  for (int i = 0; i < 4; ++i) {
    a[i]      = (__bf16)f0[i];
    a[4 + i]  = (__bf16)f1[i];
    a[8 + i]  = (__bf16)f2[i];
    a[12 + i] = (__bf16)f3[i];
  }
  return a;
}

static __device__ __forceinline__ v16bf a_frag_from_lds(
    const __bf16* __restrict__ ap) {
  v8bf lo = *(const v8bf*)ap;
  v8bf hi = *(const v8bf*)(ap + 16);
  return __builtin_shufflevector(lo, hi, 0, 1, 2, 3, 4, 5, 6, 7, 8, 9, 10, 11,
                                 12, 13, 14, 15);
}

// ---------------------------------------------------------------------------
// Utility kernels
// ---------------------------------------------------------------------------
__global__ void gn_zero(float* __restrict__ p, int n) {
  int i = blockIdx.x * blockDim.x + threadIdx.x;
  if (i < n) p[i] = 0.0f;
}

// Repack row-major f32 W[K][N] into WMMA B-fragment order (bf16):
// dst[((kt*NT+nt)*32 + lane)*16 + i] = W[kt*32 + (lane>>4)*8 + (i&7) + (i>>3)*16][nt*16 + (lane&15)]
__global__ void gn_pack_b(const float* __restrict__ W, __bf16* __restrict__ dst,
                          int K, int N) {
  int tid = blockIdx.x * blockDim.x + threadIdx.x;
  if (tid >= K * N) return;
  const int NT = N >> 4;
  int i    = tid & 15;
  int lane = (tid >> 4) & 31;
  int frag = tid >> 9;
  int nt = frag % NT;
  int kt = frag / NT;
  int k = kt * 32 + (lane >> 4) * 8 + (i & 7) + (i >> 3) * 16;
  int n = nt * 16 + (lane & 15);
  dst[tid] = (__bf16)W[(long)k * N + n];
}

// ---------------------------------------------------------------------------
// Edge MLP: concat[x[row], x[col], edge_attr, global] (256) -> 128 -> 64
// One wave per 32-edge super-tile (2 WMMA M-tiles); 4 waves per block.
// ---------------------------------------------------------------------------
__global__ __launch_bounds__(128) void gn_edge_kernel(
    const float* __restrict__ x, const int* __restrict__ ei,
    const float* __restrict__ eattr, const float* __restrict__ gattr,
    const __bf16* __restrict__ W1p, const float* __restrict__ b1,
    const __bf16* __restrict__ W2p, const float* __restrict__ b2,
    float* __restrict__ out_edge, float* __restrict__ agg,
    float* __restrict__ cnt, float* __restrict__ edge_part) {
  __shared__ __bf16 s_h1[4][MT * 16 * LAT];  // 32 KB
  __shared__ float s_esum[64];

  const int tid = threadIdx.x;
  const int wid = tid >> 5;
  const int l = tid & 31;
  const int m = l & 15;   // row within 16-tile (A/C layout)
  const int kh = l >> 4;  // K/M half select
  const int st = blockIdx.x * 4 + wid;
  const int e0 = st * (16 * MT);
  const int* rowI = ei;
  const int* colI = ei + N_EDGES;

  if (tid < 64) s_esum[tid] = 0.0f;
  __syncthreads();

  __bf16* h1 = s_h1[wid];

  // receiver-degree counts: 32 edges, one lane each
  atomicAdd(&cnt[colI[e0 + l]], 1.0f);

  // per-lane A-row base pointers
  const float* srow[MT];
  const float* drow[MT];
  const float* erow[MT];
#pragma unroll
  for (int mt = 0; mt < MT; ++mt) {
    const int e = e0 + mt * 16 + m;
    srow[mt] = x + (long)rowI[e] * FDIM;
    drow[mt] = x + (long)colI[e] * FDIM;
    erow[mt] = eattr + (long)e * FDIM;
  }

  const v8f vzero = {0.f, 0.f, 0.f, 0.f, 0.f, 0.f, 0.f, 0.f};

  // GEMM1: [32 x 256] @ [256 x 128]
  v8f acc[MT][8];
#pragma unroll
  for (int mt = 0; mt < MT; ++mt)
#pragma unroll
    for (int nt = 0; nt < 8; ++nt) acc[mt][nt] = vzero;

#pragma unroll
  for (int kt = 0; kt < 8; ++kt) {
    const int off = (kt & 1) * 32 + kh * 8;
    v16bf a[MT];
#pragma unroll
    for (int mt = 0; mt < MT; ++mt) {
      const float* base = (kt < 2)   ? srow[mt]
                          : (kt < 4) ? drow[mt]
                          : (kt < 6) ? erow[mt]
                                     : gattr;
      a[mt] = a_frag_from_f32<false>(base, off, 1.0f);
    }
#pragma unroll
    for (int nt = 0; nt < 8; ++nt) {
      v16bf b = *(const v16bf*)(W1p + ((kt * 8 + nt) * 32 + l) * 16);
#pragma unroll
      for (int mt = 0; mt < MT; ++mt)
        acc[mt][nt] = wmma_bf16(a[mt], b, acc[mt][nt]);
    }
  }

  // bias + ReLU, reshape C-layout -> row-major h1 in LDS
#pragma unroll
  for (int nt = 0; nt < 8; ++nt) {
    const float bias = b1[nt * 16 + m];
#pragma unroll
    for (int mt = 0; mt < MT; ++mt)
#pragma unroll
      for (int j = 0; j < 8; ++j) {
        float v = fmaxf(acc[mt][nt][j] + bias, 0.0f);
        h1[(mt * 16 + j + kh * 8) * LAT + nt * 16 + m] = (__bf16)v;
      }
  }
  LDS_FENCE();

  // GEMM2: [32 x 128] @ [128 x 64]
  v8f acc2[MT][4];
#pragma unroll
  for (int mt = 0; mt < MT; ++mt)
#pragma unroll
    for (int nt = 0; nt < 4; ++nt) acc2[mt][nt] = vzero;

#pragma unroll
  for (int kt = 0; kt < 4; ++kt) {
    v16bf a[MT];
#pragma unroll
    for (int mt = 0; mt < MT; ++mt)
      a[mt] = a_frag_from_lds(h1 + (mt * 16 + m) * LAT + kt * 32 + kh * 8);
#pragma unroll
    for (int nt = 0; nt < 4; ++nt) {
      v16bf b = *(const v16bf*)(W2p + ((kt * 4 + nt) * 32 + l) * 16);
#pragma unroll
      for (int mt = 0; mt < MT; ++mt)
        acc2[mt][nt] = wmma_bf16(a[mt], b, acc2[mt][nt]);
    }
  }

  // Epilogue: write new_edge, scatter into agg[col], block-partial edge sum.
#pragma unroll
  for (int nt = 0; nt < 4; ++nt) {
    const int ncol = nt * 16 + m;
    const float bias = b2[ncol];
#pragma unroll
    for (int mt = 0; mt < MT; ++mt)
#pragma unroll
      for (int j = 0; j < 8; ++j) {
        const int e = e0 + mt * 16 + j + kh * 8;
        const float v = acc2[mt][nt][j] + bias;
        out_edge[(long)e * FDIM + ncol] = v;
        atomicAdd(&agg[(long)colI[e] * FDIM + ncol], v);
        atomicAdd(&s_esum[ncol], v);
      }
  }

  __syncthreads();
  if (tid < 64) edge_part[(long)blockIdx.x * 64 + tid] = s_esum[tid];
}

// ---------------------------------------------------------------------------
// Node MLP: concat[x, agg_mean, global] (192) -> 128 -> 64
// ---------------------------------------------------------------------------
__global__ __launch_bounds__(128) void gn_node_kernel(
    const float* __restrict__ x, const float* __restrict__ gattr,
    const __bf16* __restrict__ W1p, const float* __restrict__ b1,
    const __bf16* __restrict__ W2p, const float* __restrict__ b2,
    const float* __restrict__ agg, const float* __restrict__ cnt,
    float* __restrict__ out_x, float* __restrict__ node_part) {
  __shared__ __bf16 s_h1[4][MT * 16 * LAT];
  __shared__ float s_nsum[64];

  const int tid = threadIdx.x;
  const int wid = tid >> 5;
  const int l = tid & 31;
  const int m = l & 15;
  const int kh = l >> 4;
  const int st = blockIdx.x * 4 + wid;

  if (tid < 64) s_nsum[tid] = 0.0f;
  __syncthreads();

  if (st < NODE_SUPER) {  // wave-uniform: EXEC all ones inside (WMMA req.)
    const int n0 = st * (16 * MT);
    __bf16* h1 = s_h1[wid];

    const float* xr[MT];
    const float* ar[MT];
    float inv[MT];
#pragma unroll
    for (int mt = 0; mt < MT; ++mt) {
      int n = n0 + mt * 16 + m;
      if (n > N_NODES - 1) n = N_NODES - 1;  // clamp ragged tail (stores guarded)
      xr[mt] = x + (long)n * FDIM;
      ar[mt] = agg + (long)n * FDIM;
      inv[mt] = 1.0f / fmaxf(cnt[n], 1.0f);
    }

    const v8f vzero = {0.f, 0.f, 0.f, 0.f, 0.f, 0.f, 0.f, 0.f};

    v8f acc[MT][8];
#pragma unroll
    for (int mt = 0; mt < MT; ++mt)
#pragma unroll
      for (int nt = 0; nt < 8; ++nt) acc[mt][nt] = vzero;

#pragma unroll
    for (int kt = 0; kt < 6; ++kt) {  // K = 192
      const int off = (kt & 1) * 32 + kh * 8;
      v16bf a[MT];
#pragma unroll
      for (int mt = 0; mt < MT; ++mt) {
        if (kt < 2)
          a[mt] = a_frag_from_f32<false>(xr[mt], off, 1.0f);
        else if (kt < 4)
          a[mt] = a_frag_from_f32<true>(ar[mt], off, inv[mt]);  // scatter-mean
        else
          a[mt] = a_frag_from_f32<false>(gattr, off, 1.0f);
      }
#pragma unroll
      for (int nt = 0; nt < 8; ++nt) {
        v16bf b = *(const v16bf*)(W1p + ((kt * 8 + nt) * 32 + l) * 16);
#pragma unroll
        for (int mt = 0; mt < MT; ++mt)
          acc[mt][nt] = wmma_bf16(a[mt], b, acc[mt][nt]);
      }
    }

#pragma unroll
    for (int nt = 0; nt < 8; ++nt) {
      const float bias = b1[nt * 16 + m];
#pragma unroll
      for (int mt = 0; mt < MT; ++mt)
#pragma unroll
        for (int j = 0; j < 8; ++j) {
          float v = fmaxf(acc[mt][nt][j] + bias, 0.0f);
          h1[(mt * 16 + j + kh * 8) * LAT + nt * 16 + m] = (__bf16)v;
        }
    }
    LDS_FENCE();

    v8f acc2[MT][4];
#pragma unroll
    for (int mt = 0; mt < MT; ++mt)
#pragma unroll
      for (int nt = 0; nt < 4; ++nt) acc2[mt][nt] = vzero;

#pragma unroll
    for (int kt = 0; kt < 4; ++kt) {
      v16bf a[MT];
#pragma unroll
      for (int mt = 0; mt < MT; ++mt)
        a[mt] = a_frag_from_lds(h1 + (mt * 16 + m) * LAT + kt * 32 + kh * 8);
#pragma unroll
      for (int nt = 0; nt < 4; ++nt) {
        v16bf b = *(const v16bf*)(W2p + ((kt * 4 + nt) * 32 + l) * 16);
#pragma unroll
        for (int mt = 0; mt < MT; ++mt)
          acc2[mt][nt] = wmma_bf16(a[mt], b, acc2[mt][nt]);
      }
    }

#pragma unroll
    for (int nt = 0; nt < 4; ++nt) {
      const int ncol = nt * 16 + m;
      const float bias = b2[ncol];
#pragma unroll
      for (int mt = 0; mt < MT; ++mt)
#pragma unroll
        for (int j = 0; j < 8; ++j) {
          const int n = n0 + mt * 16 + j + kh * 8;
          if (n < N_NODES) {  // guard ragged tail (no WMMA below; EXEC may diverge)
            const float v = acc2[mt][nt][j] + bias;
            out_x[(long)n * FDIM + ncol] = v;
            atomicAdd(&s_nsum[ncol], v);
          }
        }
    }
  }

  __syncthreads();
  if (tid < 64) node_part[(long)blockIdx.x * 64 + tid] = s_nsum[tid];
}

// ---------------------------------------------------------------------------
// Global MLP: reduce block partials -> concat[global, mean_x, mean_e] -> MLP
// ---------------------------------------------------------------------------
__global__ __launch_bounds__(256) void gn_global_kernel(
    const float* __restrict__ gattr, const float* __restrict__ gW1,
    const float* __restrict__ gb1, const float* __restrict__ gW2,
    const float* __restrict__ gb2, const float* __restrict__ edge_part,
    const float* __restrict__ node_part, float* __restrict__ out_g) {
  __shared__ float s_gin[192];
  __shared__ float s_h1[LAT];
  __shared__ float s_red[256];
  const int t = threadIdx.x;
  const int c = t & 63, ch = t >> 6;

  float s = 0.0f;
  for (int r = ch; r < EDGE_BLOCKS; r += 4) s += edge_part[(long)r * 64 + c];
  s_red[t] = s;
  __syncthreads();
  if (t < 64)
    s_gin[128 + t] = (s_red[t] + s_red[t + 64] + s_red[t + 128] + s_red[t + 192]) *
                     (1.0f / (float)N_EDGES);
  __syncthreads();

  s = 0.0f;
  for (int r = ch; r < NODE_BLOCKS; r += 4) s += node_part[(long)r * 64 + c];
  s_red[t] = s;
  __syncthreads();
  if (t < 64) {
    s_gin[64 + t] = (s_red[t] + s_red[t + 64] + s_red[t + 128] + s_red[t + 192]) *
                    (1.0f / (float)N_NODES);
    s_gin[t] = gattr[t];
  }
  __syncthreads();

  if (t < LAT) {
    float a = gb1[t];
    for (int k = 0; k < 192; ++k) a = fmaf(s_gin[k], gW1[k * LAT + t], a);
    s_h1[t] = fmaxf(a, 0.0f);
  }
  __syncthreads();
  if (t < 64) {
    float a = gb2[t];
    for (int k = 0; k < LAT; ++k) a = fmaf(s_h1[k], gW2[k * 64 + t], a);
    out_g[t] = a;
  }
}

// ---------------------------------------------------------------------------
extern "C" void kernel_launch(void* const* d_in, const int* in_sizes, int n_in,
                              void* d_out, int out_size, void* d_ws,
                              size_t ws_size, hipStream_t stream) {
  const float* x = (const float*)d_in[0];
  const int* ei = (const int*)d_in[1];
  const float* eattr = (const float*)d_in[2];
  const float* gattr = (const float*)d_in[3];
  const float* eW1 = (const float*)d_in[4];
  const float* eb1 = (const float*)d_in[5];
  const float* eW2 = (const float*)d_in[6];
  const float* eb2 = (const float*)d_in[7];
  const float* nW1 = (const float*)d_in[8];
  const float* nb1 = (const float*)d_in[9];
  const float* nW2 = (const float*)d_in[10];
  const float* nb2 = (const float*)d_in[11];
  const float* gW1 = (const float*)d_in[12];
  const float* gb1 = (const float*)d_in[13];
  const float* gW2 = (const float*)d_in[14];
  const float* gb2 = (const float*)d_in[15];

  float* out = (float*)d_out;
  float* out_x = out;                                // [50000*64]
  float* out_edge = out + (size_t)N_NODES * FDIM;    // [800000*64]
  float* out_g = out_edge + (size_t)N_EDGES * FDIM;  // [64]

  // Workspace layout (bytes), ~14.9 MB total:
  char* ws = (char*)d_ws;
  __bf16* W1p = (__bf16*)(ws + 0);        //  65536 B (256x128 bf16)
  __bf16* W2p = (__bf16*)(ws + 65536);    //  16384 B (128x64)
  __bf16* nW1p = (__bf16*)(ws + 81920);   //  49152 B (192x128)
  __bf16* nW2p = (__bf16*)(ws + 131072);  //  16384 B (128x64)
  float* agg = (float*)(ws + 147456);     // 12.8 MB
  float* cnt = (float*)(ws + 12947456);   // 200000 B (contiguous after agg)
  float* edge_part = (float*)(ws + 13147456);  // 6250*64*4 = 1.6 MB
  float* node_part = (float*)(ws + 14747456);  // 391*64*4 = 100096 B

  gn_pack_b<<<(256 * 128 + 255) / 256, 256, 0, stream>>>(eW1, W1p, 256, 128);
  gn_pack_b<<<(128 * 64 + 255) / 256, 256, 0, stream>>>(eW2, W2p, 128, 64);
  gn_pack_b<<<(192 * 128 + 255) / 256, 256, 0, stream>>>(nW1, nW1p, 192, 128);
  gn_pack_b<<<(128 * 64 + 255) / 256, 256, 0, stream>>>(nW2, nW2p, 128, 64);
  // zero agg + cnt (contiguous 3,250,000 floats); must re-zero every launch
  gn_zero<<<(3250000 + 255) / 256, 256, 0, stream>>>(agg, 3250000);

  gn_edge_kernel<<<EDGE_BLOCKS, 128, 0, stream>>>(x, ei, eattr, gattr, W1p, eb1,
                                                  W2p, eb2, out_edge, agg, cnt,
                                                  edge_part);
  gn_node_kernel<<<NODE_BLOCKS, 128, 0, stream>>>(x, gattr, nW1p, nb1, nW2p,
                                                  nb2, agg, cnt, out_x,
                                                  node_part);
  gn_global_kernel<<<1, 256, 0, stream>>>(gattr, gW1, gb1, gW2, gb2, edge_part,
                                          node_part, out_g);
}
